// ViWordFormer_seq_label_15109694947372
// MI455X (gfx1250) — matrix operand order repaired and checked by reading, not compiled, not run on hardware
//
#include <hip/hip_runtime.h>
#include <hip/hip_bf16.h>

#define B_    8
#define S_    512
#define D_    768
#define H_    12
#define DH_   64
#define DFF_  3072
#define OUT_  9
#define M_    (B_ * S_)   // 4096 rows

typedef __attribute__((ext_vector_type(16))) __bf16 v16bf;
typedef __attribute__((ext_vector_type(8)))  float  v8f;

union Frag16 { v16bf v; unsigned short u[16]; };

__device__ __forceinline__ unsigned short f2bf(float f) {
    unsigned int u = __float_as_uint(f);
    unsigned int r = (u + 0x7FFFu + ((u >> 16) & 1u)) >> 16;   // RNE
    return (unsigned short)r;
}

// ---------------------------------------------------------------------------
// f32 -> bf16 conversion (for weights)
// ---------------------------------------------------------------------------
__global__ void f2bf_kernel(const float* __restrict__ in, unsigned short* __restrict__ out, int n) {
    int i = blockIdx.x * 256 + threadIdx.x;
    if (i < n) out[i] = f2bf(in[i]);
}

// ---------------------------------------------------------------------------
// Embedding + sinusoidal PE + pad-mask bias
// ---------------------------------------------------------------------------
__global__ void embed_kernel(const int* __restrict__ ids, const float* __restrict__ emb,
                             float* __restrict__ xraw, float* __restrict__ maskb) {
    int m   = blockIdx.x;
    int tid = threadIdx.x;
    int id  = ids[m];
    int s   = m % S_;
    if (tid == 0) maskb[m] = (id == 0) ? -1e9f : 0.0f;
    for (int d = tid; d < D_; d += blockDim.x) {
        int   i2  = d & ~1;                                   // 2*(d/2)
        float div = __expf((float)i2 * (-9.210340371976184f / (float)D_)); // -ln(10000)/D
        float ang = (float)s * div;
        float pe  = (d & 1) ? __cosf(ang) : __sinf(ang);
        xraw[(size_t)m * D_ + d] = emb[(size_t)id * D_ + d] + pe;
    }
}

// ---------------------------------------------------------------------------
// (optional residual) + LayerNorm; writes f32 and bf16 copies
// ---------------------------------------------------------------------------
__global__ __launch_bounds__(256)
void ln_kernel(const float* __restrict__ a, const float* __restrict__ res,
               const float* __restrict__ g, const float* __restrict__ bb,
               float* __restrict__ out_f, unsigned short* __restrict__ out_bf) {
    __shared__ float red[256];
    int m   = blockIdx.x;
    int tid = threadIdx.x;
    size_t base = (size_t)m * D_;
    float v0 = a[base + tid];
    float v1 = a[base + tid + 256];
    float v2 = a[base + tid + 512];
    if (res) { v0 += res[base + tid]; v1 += res[base + tid + 256]; v2 += res[base + tid + 512]; }
    red[tid] = v0 + v1 + v2;
    __syncthreads();
    for (int o = 128; o > 0; o >>= 1) { if (tid < o) red[tid] += red[tid + o]; __syncthreads(); }
    float mean = red[0] * (1.0f / (float)D_);
    __syncthreads();
    float d0 = v0 - mean, d1 = v1 - mean, d2 = v2 - mean;
    red[tid] = d0 * d0 + d1 * d1 + d2 * d2;
    __syncthreads();
    for (int o = 128; o > 0; o >>= 1) { if (tid < o) red[tid] += red[tid + o]; __syncthreads(); }
    float rstd = rsqrtf(red[0] * (1.0f / (float)D_) + 1e-5f);
    float o0 = d0 * rstd * g[tid]       + bb[tid];
    float o1 = d1 * rstd * g[tid + 256] + bb[tid + 256];
    float o2 = d2 * rstd * g[tid + 512] + bb[tid + 512];
    out_f[base + tid]       = o0;  out_bf[base + tid]       = f2bf(o0);
    out_f[base + tid + 256] = o1;  out_bf[base + tid + 256] = f2bf(o1);
    out_f[base + tid + 512] = o2;  out_bf[base + tid + 512] = f2bf(o2);
}

// ---------------------------------------------------------------------------
// bf16 WMMA GEMM: Out[M,N] = A[M,K] @ W[K,N] + bias  (row-major)
// 128x128x32 block tile, 8 waves (2x4), each wave 64x32 = 4x2 WMMA tiles.
// Tile staging via GLOBAL_LOAD_ASYNC_TO_LDS_B128 with double buffering:
// each thread issues 4 async 16B chunks per stage (per-wave ASYNCcnt += 4);
// s_wait_asynccnt 4 waits for the current stage while the next is in flight.
// EPI: 0 = none, 1 = exact GELU.  OUTBF: 1 = bf16 out, 0 = f32 out
// ---------------------------------------------------------------------------
template <int EPI, int OUTBF>
__global__ __launch_bounds__(256)
void gemm_bf16(const unsigned short* __restrict__ A, const unsigned short* __restrict__ W,
               const float* __restrict__ bias, void* __restrict__ Out,
               int Mdim, int Ndim, int Kdim) {
    __shared__ __align__(16) unsigned short As[2][128][32];   // 8 KB per buffer
    __shared__ __align__(16) unsigned short Bs[2][32][128];   // 8 KB per buffer
    int tid  = threadIdx.x;
    int wid  = tid >> 5;
    int lane = tid & 31;
    int n0 = blockIdx.x * 128;
    int m0 = blockIdx.y * 128;
    int wm = (wid >> 2) * 64;       // 0 / 64
    int wn = (wid & 3) * 32;        // 0 / 32 / 64 / 96
    int row = lane & 15;
    int kb  = (lane >> 4) * 16;     // A-frag K split (contiguous, matches B)
    int khb = (lane >> 4) * 16;     // B-frag K split (contiguous, matches A)

    unsigned aBase = (unsigned)(unsigned long long)(&As[0][0][0]);
    unsigned bBase = (unsigned)(unsigned long long)(&Bs[0][0][0]);

    v8f acc[4][2];
    for (int mt = 0; mt < 4; mt++)
        for (int nt = 0; nt < 2; nt++)
            for (int e = 0; e < 8; e++) acc[mt][nt][e] = 0.0f;

    // Issue one stage of async tile copies (A: 512 x 16B chunks, B: 512 x 16B chunks)
    auto issue = [&](int buf, int k0) {
        for (int c = tid; c < 512; c += 256) {                 // A tile: row = c/4, 4 chunks/row
            int ra = c >> 2, ca = (c & 3) << 3;                // ca in elements (8 bf16 = 16B)
            const unsigned short* ga = A + (size_t)(m0 + ra) * Kdim + k0 + ca;
            unsigned la = aBase + buf * 8192 + ra * 64 + (ca << 1);
            asm volatile("global_load_async_to_lds_b128 %0, %1, off"
                         :: "v"(la), "v"(ga) : "memory");
        }
        for (int c = tid; c < 512; c += 256) {                 // B tile: row = c/16, 16 chunks/row
            int rb = c >> 4, cb = (c & 15) << 3;
            const unsigned short* gb = W + (size_t)(k0 + rb) * Ndim + n0 + cb;
            unsigned lb = bBase + buf * 8192 + rb * 256 + (cb << 1);
            asm volatile("global_load_async_to_lds_b128 %0, %1, off"
                         :: "v"(lb), "v"(gb) : "memory");
        }
    };

    int T = Kdim >> 5;               // number of 32-wide K steps
    issue(0, 0);
    for (int it = 0; it < T; ++it) {
        int cur = it & 1;
        if (it + 1 < T) {
            issue(cur ^ 1, (it + 1) << 5);                    // prefetch next stage
            asm volatile("s_wait_asynccnt 0x4" ::: "memory"); // current stage landed
        } else {
            asm volatile("s_wait_asynccnt 0x0" ::: "memory");
        }
        __syncthreads();

        Frag16 afr[4], bfr[2];
        for (int mt = 0; mt < 4; mt++) {
            int ar = wm + mt * 16 + row;
            for (int i = 0; i < 8; i++) {
                afr[mt].u[i]     = As[cur][ar][kb + i];
                afr[mt].u[8 + i] = As[cur][ar][kb + 8 + i];
            }
        }
        for (int nt = 0; nt < 2; nt++) {
            int bc = wn + nt * 16 + row;
            for (int i = 0; i < 16; i++) bfr[nt].u[i] = Bs[cur][khb + i][bc];
        }
        for (int mt = 0; mt < 4; mt++)
            for (int nt = 0; nt < 2; nt++)
                acc[mt][nt] = __builtin_amdgcn_wmma_f32_16x16x32_bf16(
                    false, afr[mt].v, false, bfr[nt].v, (short)0, acc[mt][nt], false, false);
        __syncthreads();                                      // all waves done reading buf cur
    }

    for (int mt = 0; mt < 4; mt++) {
        for (int nt = 0; nt < 2; nt++) {
            int   gn = n0 + wn + nt * 16 + row;
            float bv = bias ? bias[gn] : 0.0f;
            for (int r = 0; r < 8; r++) {
                int   gm = m0 + wm + mt * 16 + r + ((lane >> 4) << 3);
                float v  = acc[mt][nt][r] + bv;
                if (EPI == 1) v = 0.5f * v * (1.0f + erff(v * 0.70710678118f));
                if (OUTBF) ((unsigned short*)Out)[(size_t)gm * Ndim + gn] = f2bf(v);
                else       ((float*)Out)[(size_t)gm * Ndim + gn] = v;
            }
        }
    }
}

// ---------------------------------------------------------------------------
// scores = softmax( Q Kt / 8 + mask )  into buf[16][512]  (one wave)
// ---------------------------------------------------------------------------
__device__ __forceinline__ void scores_softmax(
    const unsigned short* __restrict__ Q, const unsigned short* __restrict__ Km,
    const float* __restrict__ maskb, int b, int h, int q0, int lane,
    float (*buf)[S_]) {
    int row = lane & 15;
    int kb  = (lane >> 4) * 16;
    int khb = (lane >> 4) * 16;

    Frag16 aq[2];
    size_t qrow = (size_t)(b * S_ + q0 + row) * D_ + h * DH_;
    for (int s = 0; s < 2; s++)
        for (int i = 0; i < 8; i++) {
            aq[s].u[i]     = Q[qrow + s * 32 + kb + i];
            aq[s].u[8 + i] = Q[qrow + s * 32 + kb + 8 + i];
        }

    for (int kt = 0; kt < S_ / 16; kt++) {
        v8f acc;
        for (int e = 0; e < 8; e++) acc[e] = 0.0f;
        size_t krow = (size_t)(b * S_ + kt * 16 + row) * D_ + h * DH_;  // key = kt*16 + (lane&15)
        for (int s = 0; s < 2; s++) {
            Frag16 bk;
            for (int i = 0; i < 16; i++) bk.u[i] = Km[krow + s * 32 + khb + i];
            acc = __builtin_amdgcn_wmma_f32_16x16x32_bf16(
                false, aq[s].v, false, bk.v, (short)0, acc, false, false);
        }
        int   n  = kt * 16 + row;
        float mb = maskb[b * S_ + n];
        for (int r = 0; r < 8; r++) {
            int mrow = r + ((lane >> 4) << 3);
            buf[mrow][n] = acc[r] * 0.125f + mb;     // 1/sqrt(64)
        }
    }
    __syncthreads();

    // softmax: 2 lanes per row, 256 cols each
    {
        int r    = lane & 15;
        int base = (lane >> 4) * 256;
        float mx = -3.0e38f;
        for (int c = 0; c < 256; c++) mx = fmaxf(mx, buf[r][base + c]);
        mx = fmaxf(mx, __shfl_xor(mx, 16, 32));
        float sum = 0.0f;
        for (int c = 0; c < 256; c++) {
            float e = __expf(buf[r][base + c] - mx);
            buf[r][base + c] = e;
            sum += e;
        }
        sum += __shfl_xor(sum, 16, 32);
        float inv = 1.0f / sum;
        for (int c = 0; c < 256; c++) buf[r][base + c] *= inv;
    }
    __syncthreads();
}

// ---------------------------------------------------------------------------
// Attention: one wave per 16-query tile per (b,h).
// attn = (softmax(raw) [+ softmax(raw_prev)]) * softmax(self); feat = attn @ V
// ---------------------------------------------------------------------------
__global__ __launch_bounds__(32)
void attn_kernel(const unsigned short* __restrict__ qs, const unsigned short* __restrict__ ks,
                 const unsigned short* __restrict__ qp, const unsigned short* __restrict__ kp,
                 const unsigned short* __restrict__ qpp, const unsigned short* __restrict__ kpp,
                 const unsigned short* __restrict__ vv, const float* __restrict__ maskb,
                 float* __restrict__ feat, int use_prev) {
    __shared__ float buf0[16][S_];
    __shared__ float buf1[16][S_];
    int lane = threadIdx.x;
    int idx  = blockIdx.x;
    int qt = idx & 31;
    int h  = (idx >> 5) % H_;
    int b  = idx / (32 * H_);
    int q0 = qt * 16;
    int row = lane & 15;
    int kb  = (lane >> 4) * 16;
    int khb = (lane >> 4) * 16;

    float* f0 = &buf0[0][0];
    float* f1 = &buf1[0][0];

    if (use_prev) {
        scores_softmax(qpp, kpp, maskb, b, h, q0, lane, buf1);   // softmax(raw_prev)
        scores_softmax(qp,  kp,  maskb, b, h, q0, lane, buf0);   // softmax(raw)
        for (int i = lane; i < 16 * S_; i += 32) f1[i] += f0[i]; // P = prev + raw
        __syncthreads();
    } else {
        scores_softmax(qp, kp, maskb, b, h, q0, lane, buf1);     // P = raw
    }
    scores_softmax(qs, ks, maskb, b, h, q0, lane, buf0);         // softmax(self)
    for (int i = lane; i < 16 * S_; i += 32) f0[i] *= f1[i];     // attn = P * self
    __syncthreads();

    // feat(16x64) = attn(16x512) @ V(512x64)
    v8f oacc[4];
    for (int nt = 0; nt < 4; nt++)
        for (int e = 0; e < 8; e++) oacc[nt][e] = 0.0f;

    for (int kk = 0; kk < S_; kk += 32) {
        Frag16 af;
        for (int i = 0; i < 8; i++) {
            af.u[i]     = f2bf(buf0[row][kk + kb + i]);
            af.u[8 + i] = f2bf(buf0[row][kk + kb + 8 + i]);
        }
        for (int nt = 0; nt < 4; nt++) {
            int col = h * DH_ + nt * 16 + row;
            Frag16 bv;
            for (int i = 0; i < 16; i++)
                bv.u[i] = vv[(size_t)(b * S_ + kk + khb + i) * D_ + col];
            oacc[nt] = __builtin_amdgcn_wmma_f32_16x16x32_bf16(
                false, af.v, false, bv.v, (short)0, oacc[nt], false, false);
        }
    }
    for (int nt = 0; nt < 4; nt++) {
        int col = h * DH_ + nt * 16 + row;
        for (int r = 0; r < 8; r++) {
            int gm = b * S_ + q0 + r + ((lane >> 4) << 3);
            feat[(size_t)gm * D_ + col] = oacc[nt][r];
        }
    }
}

// ---------------------------------------------------------------------------
// Classifier + log-softmax + per-row NLL.  One wave per row, 8 waves/block.
// ---------------------------------------------------------------------------
__global__ __launch_bounds__(256)
void cls_kernel(const float* __restrict__ x, const float* __restrict__ w,
                const float* __restrict__ bias, const int* __restrict__ labels,
                float* __restrict__ logits_out, float* __restrict__ nll) {
    int wid  = threadIdx.x >> 5;
    int lane = threadIdx.x & 31;
    int m = blockIdx.x * 8 + wid;
    float acc[OUT_];
    for (int o = 0; o < OUT_; o++) acc[o] = 0.0f;
    for (int d = lane; d < D_; d += 32) {
        float xv = x[(size_t)m * D_ + d];
        for (int o = 0; o < OUT_; o++) acc[o] += xv * w[d * OUT_ + o];
    }
    for (int off = 16; off > 0; off >>= 1)
        for (int o = 0; o < OUT_; o++) acc[o] += __shfl_xor(acc[o], off, 32);
    if (lane == 0) {
        float mx = -3.0e38f;
        for (int o = 0; o < OUT_; o++) { acc[o] += bias[o]; mx = fmaxf(mx, acc[o]); }
        float sum = 0.0f;
        for (int o = 0; o < OUT_; o++) sum += __expf(acc[o] - mx);
        float lse = mx + __logf(sum);
        for (int o = 0; o < OUT_; o++) logits_out[(size_t)m * OUT_ + o] = acc[o];
        nll[m] = lse - acc[labels[m]];
    }
}

__global__ __launch_bounds__(256)
void loss_kernel(const float* __restrict__ nll, float* __restrict__ out) {
    __shared__ float red[256];
    float s = 0.0f;
    for (int i = threadIdx.x; i < M_; i += 256) s += nll[i];
    red[threadIdx.x] = s;
    __syncthreads();
    for (int o = 128; o > 0; o >>= 1) {
        if (threadIdx.x < o) red[threadIdx.x] += red[threadIdx.x + o];
        __syncthreads();
    }
    if (threadIdx.x == 0) out[0] = red[0] * (1.0f / (float)M_);
}

// ---------------------------------------------------------------------------
// Host orchestration
// ---------------------------------------------------------------------------
extern "C" void kernel_launch(void* const* d_in, const int* in_sizes, int n_in,
                              void* d_out, int out_size, void* d_ws, size_t ws_size,
                              hipStream_t stream) {
    (void)in_sizes; (void)n_in; (void)out_size; (void)ws_size;

    const int*   ids    = (const int*)d_in[0];
    const int*   labels = (const int*)d_in[1];
    const float* emb    = (const float*)d_in[2];
    const float* ln0_g  = (const float*)d_in[3];
    const float* ln0_b  = (const float*)d_in[4];
    const float* wq_s   = (const float*)d_in[5];
    const float* bq_s   = (const float*)d_in[6];
    const float* wk_s   = (const float*)d_in[7];
    const float* bk_s   = (const float*)d_in[8];
    const float* wq_p   = (const float*)d_in[9];
    const float* bq_p   = (const float*)d_in[10];
    const float* wk_p   = (const float*)d_in[11];
    const float* bk_p   = (const float*)d_in[12];
    const float* w_out  = (const float*)d_in[13];
    const float* b_out  = (const float*)d_in[14];
    const float* w_ff1  = (const float*)d_in[15];
    const float* b_ff1  = (const float*)d_in[16];
    const float* w_ff2  = (const float*)d_in[17];
    const float* b_ff2  = (const float*)d_in[18];
    const float* ln1_g  = (const float*)d_in[19];
    const float* ln1_b  = (const float*)d_in[20];
    const float* ln2_g  = (const float*)d_in[21];
    const float* ln2_b  = (const float*)d_in[22];
    const float* w_cls  = (const float*)d_in[23];
    const float* b_cls  = (const float*)d_in[24];

    char* p = (char*)d_ws;
    auto alloc = [&](size_t bytes) -> void* {
        void* r = (void*)p;
        p += (bytes + 255) & ~(size_t)255;
        return r;
    };
    float*          xraw  = (float*)alloc((size_t)M_ * D_ * 4);
    float*          x     = (float*)alloc((size_t)M_ * D_ * 4);
    unsigned short* x_bf  = (unsigned short*)alloc((size_t)M_ * D_ * 2);
    float*          maskb = (float*)alloc((size_t)M_ * 4);
    unsigned short* qsb   = (unsigned short*)alloc((size_t)M_ * D_ * 2);
    unsigned short* ksb   = (unsigned short*)alloc((size_t)M_ * D_ * 2);
    unsigned short* qp0   = (unsigned short*)alloc((size_t)M_ * D_ * 2);
    unsigned short* kp0   = (unsigned short*)alloc((size_t)M_ * D_ * 2);
    unsigned short* qp1   = (unsigned short*)alloc((size_t)M_ * D_ * 2);
    unsigned short* kp1   = (unsigned short*)alloc((size_t)M_ * D_ * 2);
    unsigned short* vb    = (unsigned short*)alloc((size_t)M_ * D_ * 2);
    float*          feat  = (float*)alloc((size_t)M_ * D_ * 4);
    unsigned short* h1    = (unsigned short*)alloc((size_t)M_ * DFF_ * 2);
    float*          ffo   = (float*)alloc((size_t)M_ * D_ * 4);
    float*          nllb  = (float*)alloc((size_t)M_ * 4);
    unsigned short* wqs_b = (unsigned short*)alloc((size_t)D_ * D_ * 2);
    unsigned short* wks_b = (unsigned short*)alloc((size_t)D_ * D_ * 2);
    unsigned short* wqp_b = (unsigned short*)alloc((size_t)D_ * D_ * 2);
    unsigned short* wkp_b = (unsigned short*)alloc((size_t)D_ * D_ * 2);
    unsigned short* wo_b  = (unsigned short*)alloc((size_t)D_ * D_ * 2);
    unsigned short* wf1_b = (unsigned short*)alloc((size_t)D_ * DFF_ * 2);
    unsigned short* wf2_b = (unsigned short*)alloc((size_t)DFF_ * D_ * 2);

    auto conv = [&](const float* w, unsigned short* o, int n) {
        f2bf_kernel<<<(n + 255) / 256, 256, 0, stream>>>(w, o, n);
    };
    conv(wq_s, wqs_b, D_ * D_);
    conv(wk_s, wks_b, D_ * D_);
    conv(wq_p, wqp_b, D_ * D_);
    conv(wk_p, wkp_b, D_ * D_);
    conv(w_out, wo_b, D_ * D_);
    conv(w_ff1, wf1_b, D_ * DFF_);
    conv(w_ff2, wf2_b, DFF_ * D_);

    embed_kernel<<<M_, 256, 0, stream>>>(ids, emb, xraw, maskb);
    ln_kernel<<<M_, 256, 0, stream>>>(xraw, nullptr, ln0_g, ln0_b, x, x_bf);

    dim3 gD(D_ / 128, M_ / 128);      // projections: N=768
    dim3 gF1(DFF_ / 128, M_ / 128);   // FF1: N=3072
    for (int l = 0; l < 2; l++) {
        unsigned short* qpl = l ? qp1 : qp0;
        unsigned short* kpl = l ? kp1 : kp0;
        gemm_bf16<0, 1><<<gD, 256, 0, stream>>>(x_bf, wqs_b, bq_s, qsb, M_, D_, D_);
        gemm_bf16<0, 1><<<gD, 256, 0, stream>>>(x_bf, wks_b, bk_s, ksb, M_, D_, D_);
        gemm_bf16<0, 1><<<gD, 256, 0, stream>>>(x_bf, wqp_b, bq_p, qpl, M_, D_, D_);
        gemm_bf16<0, 1><<<gD, 256, 0, stream>>>(x_bf, wkp_b, bk_p, kpl, M_, D_, D_);
        gemm_bf16<0, 1><<<gD, 256, 0, stream>>>(x_bf, wo_b,  b_out, vb, M_, D_, D_);

        attn_kernel<<<B_ * H_ * (S_ / 16), 32, 0, stream>>>(
            qsb, ksb, qpl, kpl, qp0, kp0, vb, maskb, feat, l);

        ln_kernel<<<M_, 256, 0, stream>>>(feat, x, ln1_g, ln1_b, x, x_bf);

        gemm_bf16<1, 1><<<gF1, 256, 0, stream>>>(x_bf, wf1_b, b_ff1, h1, M_, DFF_, D_);
        gemm_bf16<0, 0><<<gD,  256, 0, stream>>>(h1,   wf2_b, b_ff2, ffo, M_, D_, DFF_);

        ln_kernel<<<M_, 256, 0, stream>>>(ffo, x, ln2_g, ln2_b, x, x_bf);
    }

    float* logits = (float*)d_out;
    cls_kernel<<<M_ / 8, 256, 0, stream>>>(x, w_cls, b_cls, labels, logits, nllb);
    loss_kernel<<<1, 256, 0, stream>>>(nllb, logits + (size_t)M_ * OUT_);
}
